// MultiHeadAttention_64750926955125
// MI455X (gfx1250) — compile-verified, matrix operand order checked by reading, not run on hardware
//
#include <hip/hip_runtime.h>
#include <hip/hip_bf16.h>
#include <stdint.h>

// MHA forward for MI455X (gfx1250), bf16 WMMA pipeline with f32 accumulation.
// out (8,512,1024) f32  ++  attn (16,8,512,512) f32 concatenated in d_out.
//
// GEMMs: 128x128 block tile, 8 waves, each wave 32x64 (2x4 WMMA tiles),
// K-step 64, double-buffered LDS staged via GLOBAL_LOAD_ASYNC_TO_LDS_B128
// (ASYNCcnt) so HBM->LDS copies overlap the WMMA stream.

#define N_HEADS 16
#define D_MODEL 1024
#define D_KEY   64
#define BATCH   8
#define TSEQ    512
#define M_ALL   (BATCH * TSEQ)        // 4096
#define KQ_N    (2 * N_HEADS * D_KEY) // 2048
#define FC_K    (N_HEADS * D_MODEL)   // 16384
#define ATT_SCALE 0.125f              // 64^-0.5

typedef __attribute__((ext_vector_type(16))) __bf16 v16bf;
typedef __attribute__((ext_vector_type(8)))  float  v8f;

struct Frag { union { v16bf v; uint32_t u[8]; }; };

static __device__ __forceinline__ unsigned short f32_to_bf16(float f) {
  uint32_t u = __float_as_uint(f);
  u += 0x7FFFu + ((u >> 16) & 1u);   // round-to-nearest-even
  return (unsigned short)(u >> 16);
}

// Async 16B copy global -> LDS (tracked by ASYNCcnt, no VGPR round-trip).
// Flat pointers to LDS carry the LDS byte address in their low 32 bits.
static __device__ __forceinline__ void async_copy_b128(void* lds_dst, const void* gsrc) {
  uint32_t lds_addr = (uint32_t)(uintptr_t)lds_dst;
  asm volatile("global_load_async_to_lds_b128 %0, %1, off"
               :: "v"(lds_addr), "v"(gsrc)
               : "memory");
}
static __device__ __forceinline__ void wait_async0() {
  asm volatile("s_wait_asynccnt 0x0" ::: "memory");
}

// Load a 16x32 bf16 A-fragment (or Bt-fragment) from a [rows][ld] bf16 matrix.
// ISA 16-bit A layout: lanes 0-15 -> M=lane, K-base 0; lanes 16-31 -> M=lane-16, K-base 8.
// VGPR v=0..3 holds K = base+2v, base+2v+1; v=4..7 holds K = base+16+2(v-4)..
static __device__ __forceinline__ void load_frag(Frag& f, const unsigned short* base,
                                                 int ld, int row0, int k0) {
  int lane = threadIdx.x & 31;
  const unsigned short* p =
      base + (size_t)(row0 + (lane & 15)) * ld + k0 + ((lane >> 4) << 3);
  f.u[0] = *(const uint32_t*)(p + 0);
  f.u[1] = *(const uint32_t*)(p + 2);
  f.u[2] = *(const uint32_t*)(p + 4);
  f.u[3] = *(const uint32_t*)(p + 6);
  f.u[4] = *(const uint32_t*)(p + 16);
  f.u[5] = *(const uint32_t*)(p + 18);
  f.u[6] = *(const uint32_t*)(p + 20);
  f.u[7] = *(const uint32_t*)(p + 22);
}

static __device__ __forceinline__ v8f wmma_bf16(const Frag& a, const Frag& b, v8f c) {
  return __builtin_amdgcn_wmma_f32_16x16x32_bf16(false, a.v, false, b.v,
                                                 (short)0, c, false, false);
}

// 256 threads stage a 128x64 bf16 tile with async LDS copies (4 x 16B each).
static __device__ __forceinline__ void stage128x64(unsigned short* dst,
                                                   const unsigned short* src,
                                                   int ld, int row0, int k0) {
  int t = threadIdx.x;
  int c = (t & 7) << 3;    // col chunk 0,8,..,56
  int r0 = t >> 3;         // 0..31
#pragma unroll
  for (int i = 0; i < 4; ++i) {
    int r = r0 + 32 * i;
    async_copy_b128(dst + r * 64 + c, src + (size_t)(row0 + r) * ld + k0 + c);
  }
}

// ---------------------------------------------------------------- casts
__global__ void cast_bf16_kernel(const float* __restrict__ src,
                                 unsigned short* __restrict__ dst, int n) {
  int i = blockIdx.x * blockDim.x + threadIdx.x;
  int stride = gridDim.x * blockDim.x;
  for (; i < n; i += stride) dst[i] = f32_to_bf16(src[i]);
}

// Vt[b][c][k] = v[b][k][c]  (bf16), so P@V can read Bt = Vt row-major.
__global__ void transpose_v_kernel(const float* __restrict__ v,
                                   unsigned short* __restrict__ vt) {
  int i = blockIdx.x * blockDim.x + threadIdx.x;
  int stride = gridDim.x * blockDim.x;
  for (; i < M_ALL * D_MODEL; i += stride) {
    int b = i >> 19;          // 512*1024 = 2^19
    int r = i & 524287;
    int k = r >> 10;
    int c = r & 1023;
    vt[(b << 19) + c * TSEQ + k] = f32_to_bf16(v[i]);
  }
}

// -------------------------------------------------- shared GEMM wave core
// Each wave: rows [wm, wm+32), cols [wn, wn+64) of the 128x128 block tile.
// Performs 16 WMMAs per staged K-step of 64.
#define GEMM_WAVE_DECL()                                   \
  int lane = threadIdx.x & 31, w = threadIdx.x >> 5;       \
  int wm = (w >> 1) * 32, wn = (w & 1) * 64;               \
  v8f acc[2][4] = {};

#define GEMM_WAVE_MMA(As, Bs)                              \
  _Pragma("unroll")                                        \
  for (int ks = 0; ks < 64; ks += 32) {                    \
    Frag a0, a1, b0, b1, b2, b3;                           \
    load_frag(a0, As, 64, wm, ks);                         \
    load_frag(a1, As, 64, wm + 16, ks);                    \
    load_frag(b0, Bs, 64, wn, ks);                         \
    load_frag(b1, Bs, 64, wn + 16, ks);                    \
    load_frag(b2, Bs, 64, wn + 32, ks);                    \
    load_frag(b3, Bs, 64, wn + 48, ks);                    \
    acc[0][0] = wmma_bf16(a0, b0, acc[0][0]);              \
    acc[0][1] = wmma_bf16(a0, b1, acc[0][1]);              \
    acc[0][2] = wmma_bf16(a0, b2, acc[0][2]);              \
    acc[0][3] = wmma_bf16(a0, b3, acc[0][3]);              \
    acc[1][0] = wmma_bf16(a1, b0, acc[1][0]);              \
    acc[1][1] = wmma_bf16(a1, b1, acc[1][1]);              \
    acc[1][2] = wmma_bf16(a1, b2, acc[1][2]);              \
    acc[1][3] = wmma_bf16(a1, b3, acc[1][3]);              \
  }

// --------------------------------------------- GEMM1: kq = v @ Wkq^T + b
__global__ __launch_bounds__(256)
void kq_gemm_kernel(const unsigned short* __restrict__ A,    // [4096][1024]
                    const unsigned short* __restrict__ BT,   // Wkq [2048][1024]
                    const float* __restrict__ bias,
                    unsigned short* __restrict__ Kbuf,       // [16][4096][64]
                    unsigned short* __restrict__ Qbuf) {     // [16][4096][64]
  __shared__ __align__(16) unsigned short As[2][128 * 64];
  __shared__ __align__(16) unsigned short Bs[2][128 * 64];
  int m0 = blockIdx.x * 128, n0 = blockIdx.y * 128;
  GEMM_WAVE_DECL();
  stage128x64(As[0], A, D_MODEL, m0, 0);
  stage128x64(Bs[0], BT, D_MODEL, n0, 0);
  int cur = 0;
  for (int k0 = 0; k0 < D_MODEL; k0 += 64) {
    wait_async0();
    __syncthreads();
    int kn = k0 + 64;
    if (kn < D_MODEL) {               // prefetch next tile while computing this one
      stage128x64(As[cur ^ 1], A, D_MODEL, m0, kn);
      stage128x64(Bs[cur ^ 1], BT, D_MODEL, n0, kn);
    }
    GEMM_WAVE_MMA(As[cur], Bs[cur]);
    cur ^= 1;
  }
#pragma unroll
  for (int i = 0; i < 2; ++i)
#pragma unroll
    for (int j = 0; j < 4; ++j) {
      int n = n0 + wn + j * 16 + (lane & 15);
      int head = n >> 7, r = n & 127;   // channel layout (head, 2*dk): k=r<64, q=r>=64
      float bv = bias[n];
      for (int rr = 0; rr < 8; ++rr) {
        int m = m0 + wm + i * 16 + rr + ((lane >> 4) << 3);
        unsigned short val = f32_to_bf16(acc[i][j][rr] + bv);
        if (r < D_KEY) Kbuf[((size_t)head * M_ALL + m) * D_KEY + r] = val;
        else           Qbuf[((size_t)head * M_ALL + m) * D_KEY + (r - D_KEY)] = val;
      }
    }
}

// -------------------- fused attention: S = scale*QK^T, mask, softmax
__global__ __launch_bounds__(256)
void attn_kernel(const unsigned short* __restrict__ Qbuf,
                 const unsigned short* __restrict__ Kbuf,
                 const unsigned char* __restrict__ mask,    // [8][512][512] bool
                 float* __restrict__ attn_out,              // [16][8][512][512]
                 unsigned short* __restrict__ Pbuf) {       // same shape, bf16
  __shared__ __align__(16) float S[16][528];                // pad 16 -> conflict-free softmax
  int hb = blockIdx.x;                 // h*8 + b
  int qt = blockIdx.y;                 // 16-row q tile
  int h = hb >> 3, bi = hb & 7;
  int lane = threadIdx.x & 31, w = threadIdx.x >> 5;
  const unsigned short* Qh = Qbuf + (size_t)h * M_ALL * D_KEY;
  const unsigned short* Kh = Kbuf + (size_t)h * M_ALL * D_KEY;
  int qrow0 = bi * TSEQ + qt * 16;

  Frag qa0, qa1;
  load_frag(qa0, Qh, D_KEY, qrow0, 0);
  load_frag(qa1, Qh, D_KEY, qrow0, 32);

  v8f acc[4] = {};
#pragma unroll
  for (int t = 0; t < 4; ++t) {
    int n0 = w * 64 + t * 16;          // key-column tile for this wave
    Frag kb0, kb1;
    load_frag(kb0, Kh, D_KEY, bi * TSEQ + n0, 0);
    load_frag(kb1, Kh, D_KEY, bi * TSEQ + n0, 32);
    acc[t] = wmma_bf16(qa0, kb0, acc[t]);
    acc[t] = wmma_bf16(qa1, kb1, acc[t]);
  }
#pragma unroll
  for (int t = 0; t < 4; ++t) {
    int col = w * 64 + t * 16 + (lane & 15);
    for (int rr = 0; rr < 8; ++rr) {
      int row = rr + ((lane >> 4) << 3);
      int q = qt * 16 + row;
      float sv = acc[t][rr] * ATT_SCALE;
      if (mask[((size_t)bi * TSEQ + q) * TSEQ + col]) sv = -100000.0f;
      S[row][col] = sv;
    }
  }
  __syncthreads();

  // softmax: 16 lanes per row, 32 cols per lane; xor-shuffles stay inside each half-wave
  int row = threadIdx.x >> 4;
  int lr = threadIdx.x & 15;
  float vals[32];
  float mx = -3.0e38f;
  for (int i = 0; i < 32; ++i) { vals[i] = S[row][lr + 16 * i]; mx = fmaxf(mx, vals[i]); }
  for (int d = 1; d < 16; d <<= 1) mx = fmaxf(mx, __shfl_xor(mx, d, 32));
  float sum = 0.0f;
  for (int i = 0; i < 32; ++i) { vals[i] = __expf(vals[i] - mx); sum += vals[i]; }
  for (int d = 1; d < 16; d <<= 1) sum += __shfl_xor(sum, d, 32);
  float inv = 1.0f / sum;
  int q = qt * 16 + row;
  size_t base = ((size_t)hb * TSEQ + q) * TSEQ;
  for (int i = 0; i < 32; ++i) {
    float p = vals[i] * inv;
    int col = lr + 16 * i;
    attn_out[base + col] = p;
    Pbuf[base + col] = f32_to_bf16(p);
  }
}

// --------------------------------------------- GEMM3: HV[hb] = P[hb] @ V[b]
__global__ __launch_bounds__(256)
void pv_gemm_kernel(const unsigned short* __restrict__ Pbuf,  // [128][512][512]
                    const unsigned short* __restrict__ Vt,    // [8][1024][512]
                    unsigned short* __restrict__ HV) {        // [128][512][1024]
  __shared__ __align__(16) unsigned short As[2][128 * 64];
  __shared__ __align__(16) unsigned short Bs[2][128 * 64];
  int hb = blockIdx.z;
  int m0 = blockIdx.y * 128, n0 = blockIdx.x * 128;
  const unsigned short* A  = Pbuf + (size_t)hb * TSEQ * TSEQ;
  const unsigned short* BT = Vt + (size_t)(hb & 7) * D_MODEL * TSEQ;
  GEMM_WAVE_DECL();
  stage128x64(As[0], A, TSEQ, m0, 0);
  stage128x64(Bs[0], BT, TSEQ, n0, 0);
  int cur = 0;
  for (int k0 = 0; k0 < TSEQ; k0 += 64) {
    wait_async0();
    __syncthreads();
    int kn = k0 + 64;
    if (kn < TSEQ) {
      stage128x64(As[cur ^ 1], A, TSEQ, m0, kn);
      stage128x64(Bs[cur ^ 1], BT, TSEQ, n0, kn);
    }
    GEMM_WAVE_MMA(As[cur], Bs[cur]);
    cur ^= 1;
  }
#pragma unroll
  for (int i = 0; i < 2; ++i)
#pragma unroll
    for (int j = 0; j < 4; ++j) {
      int n = n0 + wn + j * 16 + (lane & 15);
      for (int rr = 0; rr < 8; ++rr) {
        int m = m0 + wm + i * 16 + rr + ((lane >> 4) << 3);
        HV[((size_t)hb * TSEQ + m) * D_MODEL + n] = f32_to_bf16(acc[i][j][rr]);
      }
    }
}

// -------------------- GEMM4: out = concat_heads(HV) @ fc_w^T + fc_b
__global__ __launch_bounds__(256)
void fc_gemm_kernel(const unsigned short* __restrict__ HV,   // [16][4096][1024] head-major
                    const unsigned short* __restrict__ FCW,  // [1024][16384]
                    const float* __restrict__ bias,
                    float* __restrict__ out) {               // [4096][1024]
  __shared__ __align__(16) unsigned short As[2][128 * 64];
  __shared__ __align__(16) unsigned short Bs[2][128 * 64];
  int m0 = blockIdx.y * 128, n0 = blockIdx.x * 128;
  GEMM_WAVE_DECL();
  int t = threadIdx.x;
  int sc = (t & 7) << 3;
  int sr0 = t >> 3;
  // A row m, reduction index jk = h*1024+cp lives at HV[(h*4096 + m)*1024 + cp];
  // a 64-wide k tile never crosses a head boundary (64 | 1024).
#define FC_STAGE_A(buf, kk)                                                   \
  do {                                                                        \
    int hh = (kk) >> 10;                                                      \
    int cp = ((kk) & 1023) + sc;                                              \
    _Pragma("unroll")                                                         \
    for (int i = 0; i < 4; ++i) {                                             \
      int r = sr0 + 32 * i;                                                   \
      async_copy_b128((buf) + r * 64 + sc,                                    \
                      HV + ((size_t)hh * M_ALL + (m0 + r)) * D_MODEL + cp);   \
    }                                                                         \
  } while (0)

  FC_STAGE_A(As[0], 0);
  stage128x64(Bs[0], FCW, FC_K, n0, 0);
  int cur = 0;
  for (int k0 = 0; k0 < FC_K; k0 += 64) {
    wait_async0();
    __syncthreads();
    int kn = k0 + 64;
    if (kn < FC_K) {
      FC_STAGE_A(As[cur ^ 1], kn);
      stage128x64(Bs[cur ^ 1], FCW, FC_K, n0, kn);
    }
    GEMM_WAVE_MMA(As[cur], Bs[cur]);
    cur ^= 1;
  }
#undef FC_STAGE_A
#pragma unroll
  for (int i = 0; i < 2; ++i)
#pragma unroll
    for (int j = 0; j < 4; ++j) {
      int n = n0 + wn + j * 16 + (lane & 15);
      float bv = bias[n];
      for (int rr = 0; rr < 8; ++rr) {
        int m = m0 + wm + i * 16 + rr + ((lane >> 4) << 3);
        out[(size_t)m * D_MODEL + n] = acc[i][j][rr] + bv;
      }
    }
}

extern "C" void kernel_launch(void* const* d_in, const int* in_sizes, int n_in,
                              void* d_out, int out_size, void* d_ws, size_t ws_size,
                              hipStream_t stream) {
  (void)in_sizes; (void)n_in; (void)out_size; (void)ws_size;
  const float*         v      = (const float*)d_in[0];
  const unsigned char* mask   = (const unsigned char*)d_in[1];  // bool array (1B)
  const float*         wkq    = (const float*)d_in[2];
  const float*         wkq_b  = (const float*)d_in[3];
  const float*         fcw    = (const float*)d_in[4];
  const float*         fcb    = (const float*)d_in[5];
  float* out      = (float*)d_out;
  float* attn_out = out + (size_t)M_ALL * D_MODEL;   // tuple order: (out, attn)

  char* ws = (char*)d_ws;
  size_t off = 0;
  auto alloc = [&](size_t bytes) -> void* {
    void* p = ws + off;
    off += (bytes + 255) & ~(size_t)255;
    return p;
  };
  unsigned short* Vb   = (unsigned short*)alloc((size_t)M_ALL * D_MODEL * 2);
  unsigned short* Vt   = (unsigned short*)alloc((size_t)M_ALL * D_MODEL * 2);
  unsigned short* Wb   = (unsigned short*)alloc((size_t)KQ_N * D_MODEL * 2);
  unsigned short* FCWb = (unsigned short*)alloc((size_t)D_MODEL * FC_K * 2);
  unsigned short* Kb   = (unsigned short*)alloc((size_t)N_HEADS * M_ALL * D_KEY * 2);
  unsigned short* Qb   = (unsigned short*)alloc((size_t)N_HEADS * M_ALL * D_KEY * 2);
  unsigned short* Pb   = (unsigned short*)alloc((size_t)N_HEADS * BATCH * TSEQ * TSEQ * 2);
  unsigned short* HV   = (unsigned short*)alloc((size_t)N_HEADS * M_ALL * D_MODEL * 2);

  cast_bf16_kernel<<<2048, 256, 0, stream>>>(v, Vb, M_ALL * D_MODEL);
  transpose_v_kernel<<<2048, 256, 0, stream>>>(v, Vt);
  cast_bf16_kernel<<<1024, 256, 0, stream>>>(wkq, Wb, KQ_N * D_MODEL);
  cast_bf16_kernel<<<4096, 256, 0, stream>>>(fcw, FCWb, D_MODEL * FC_K);

  kq_gemm_kernel<<<dim3(M_ALL / 128, KQ_N / 128), 256, 0, stream>>>(Vb, Wb, wkq_b, Kb, Qb);
  attn_kernel<<<dim3(N_HEADS * BATCH, TSEQ / 16), 256, 0, stream>>>(Qb, Kb, mask, attn_out, Pb);
  pv_gemm_kernel<<<dim3(D_MODEL / 128, TSEQ / 128, N_HEADS * BATCH), 256, 0, stream>>>(Pb, Vt, HV);
  fc_gemm_kernel<<<dim3(D_MODEL / 128, M_ALL / 128), 256, 0, stream>>>(HV, FCWb, fcb, out);
}
